// GeneAttentionLayer_50543175139786
// MI455X (gfx1250) — matrix operand order repaired and checked by reading, not compiled, not run on hardware
//
#include <hip/hip_runtime.h>
#include <math.h>

typedef __attribute__((ext_vector_type(16))) __bf16 v16bf;
typedef __attribute__((ext_vector_type(8)))  __bf16 v8bf;
typedef __attribute__((ext_vector_type(8)))  float  v8f;
typedef __attribute__((ext_vector_type(4)))  float  v4f;

#define GG 2048
#define DD 256
#define KT 32                   // keys per iteration
#define NT (GG / KT)            // 64 key tiles
#define WAVES 8
#define ROWS_PER_WG (16 * WAVES)
#define KPITCH 272              // padded bf16 pitch for K rows (17*16 halfs)
#define KTILE_BYTES (KT * KPITCH * 2)        // 17408
#define VTILE_BYTES (KT * DD * 2)            // 16384

// ---------------- prep: f32 -> bf16, K padded row-major, V pre-swizzled to WMMA B-layout ---------
__global__ void prep_kv(const float* __restrict__ Kg, const float* __restrict__ Vg,
                        __bf16* __restrict__ Kbf, __bf16* __restrict__ Vbf)
{
    const size_t i4  = ((size_t)blockIdx.x * blockDim.x + threadIdx.x) * 4;
    const int    d   = (int)(i4 & 255);
    const size_t row = i4 >> 8;               // b*G + k
    const int    k   = (int)(row & (GG - 1));

    const v4f kv = *(const v4f*)(Kg + i4);
    const v4f vv = *(const v4f*)(Vg + i4);

    __bf16* kd = Kbf + row * KPITCH + d;
    #pragma unroll
    for (int j = 0; j < 4; ++j) kd[j] = (__bf16)kv[j];

    const int    tile = k >> 5, r = k & 31, g2 = r >> 4, e = r & 15;
    const size_t vb   = (((row >> 11) << 6) + tile) * (size_t)8192;   // (b*64 + tile)*8192 halfs
    #pragma unroll
    for (int j = 0; j < 4; ++j) {
        const int dd = d + j;
        const int tt = dd >> 4;
        const int ln = (dd & 15) + 16 * g2;
        Vbf[vb + ((tt * 32 + ln) << 4) + e] = (__bf16)vv[j];
    }
}

// ---------------- CDNA5 async copy: 16B per lane, global -> LDS --------------------------------
__device__ __forceinline__ void async_ld16(unsigned lds_addr, const void* gptr) {
    asm volatile("global_load_async_to_lds_b128 %0, %1, off"
                 :: "v"(lds_addr), "v"(gptr) : "memory");
}

// waves 0/1 issue one tile's worth of async copies (no wait here)
__device__ __forceinline__ void stage_issue(int wv, int lane,
                                            const __bf16* __restrict__ Kbf,
                                            const __bf16* __restrict__ Vbf,
                                            int b, int kb,
                                            unsigned dstK, unsigned dstV)
{
    if (wv == 0) {
        const char* src = (const char*)(Kbf + (size_t)(b * GG + kb) * KPITCH) + lane * 16;
        const unsigned dst = dstK + lane * 16;
        #pragma unroll
        for (int i = 0; i < KTILE_BYTES / 512; ++i)        // 34 x 512B
            async_ld16(dst + i * 512, src + i * 512);
    } else if (wv == 1) {
        const char* src = (const char*)(Vbf + ((size_t)(b << 6) + (kb >> 5)) * 8192) + lane * 16;
        const unsigned dst = dstV + lane * 16;
        #pragma unroll
        for (int i = 0; i < VTILE_BYTES / 512; ++i)        // 32 x 512B
            async_ld16(dst + i * 512, src + i * 512);
    }
}

// ---------------- fused gated-attention kernel -------------------------------------------------
__global__ __launch_bounds__(256, 1)
void gene_attn_fused(const float* __restrict__ Qg, const __bf16* __restrict__ Kbf,
                     const __bf16* __restrict__ Vbf, const float* __restrict__ Mg,
                     float* __restrict__ Og)
{
    __shared__ __attribute__((aligned(32))) __bf16 ldsK[2][KT * KPITCH];     // double-buffered K
    __shared__ __attribute__((aligned(32))) __bf16 ldsVb[2][16 * 32 * 16];   // double-buffered V (B-layout)
    __shared__ __attribute__((aligned(32))) __bf16 ldsP[WAVES * 16 * 32];    // per-wave P transpose

    const int tid  = threadIdx.x;
    const int wv   = tid >> 5;
    const int lane = tid & 31;
    const int ln16 = lane & 15;
    const int grp  = lane >> 4;
    const int b    = blockIdx.y;
    const int q0   = blockIdx.x * ROWS_PER_WG + wv * 16;

    const float* Qb = Qg + ((size_t)b * GG) * DD;
    const float* Mb = Mg + ((size_t)b * GG) * (size_t)GG;
    float*       Ob = Og + ((size_t)b * GG) * DD;

    const float lscale = 0.0625f * 1.44269504088896f;   // 1/sqrt(256)*log2(e), folded into Q

    // ---- Q tile (16x256) resident in registers, A-layout, scale folded in ----
    v16bf aq[8];
    {
        const float* qrow = Qb + (size_t)(q0 + ln16) * DD;
        #pragma unroll
        for (int c = 0; c < 8; ++c) {
            const int d0 = 32 * c + 16 * grp;
            #pragma unroll
            for (int e = 0; e < 8; ++e) {
                aq[c][e]     = (__bf16)(qrow[d0 + e]      * lscale);
                aq[c][8 + e] = (__bf16)(qrow[d0 + 8 + e] * lscale);
            }
        }
    }

    v8f acc[16];
    #pragma unroll
    for (int t = 0; t < 16; ++t) acc[t] = v8f{};

    float rowmax[8], Zs[8], Ss[8];
    #pragma unroll
    for (int r = 0; r < 8; ++r) { rowmax[r] = -__builtin_inff(); Zs[r] = 0.f; Ss[r] = 0.f; }

    const unsigned kbase[2] = { (unsigned)(size_t)&ldsK[0][0],  (unsigned)(size_t)&ldsK[1][0]  };
    const unsigned vbase[2] = { (unsigned)(size_t)&ldsVb[0][0], (unsigned)(size_t)&ldsVb[1][0] };

    // prologue: issue tile 0 into buffer 0
    stage_issue(wv, lane, Kbf, Vbf, b, 0, kbase[0], vbase[0]);

    for (int it = 0; it < NT; ++it) {
        const int kb  = it * KT;
        const int buf = it & 1;

        if (wv <= 1) asm volatile("s_wait_asynccnt 0x0" ::: "memory");
        __syncthreads();                     // buf now valid for everyone; prior reads retired

        if (it + 1 < NT)                     // overlap next tile's copy with this tile's compute
            stage_issue(wv, lane, Kbf, Vbf, b, kb + KT, kbase[buf ^ 1], vbase[buf ^ 1]);

        const __bf16* tK = &ldsK[buf][0];
        const __bf16* tV = &ldsVb[buf][0];

        // ---- logits: batch all B-tiles, then dependency-free WMMA chain ----
        v8f lt[2];
        #pragma unroll
        for (int h = 0; h < 2; ++h) {
            v16bf bks[8];
            #pragma unroll
            for (int c = 0; c < 8; ++c)
                bks[c] = *(const v16bf*)&tK[(16 * h + ln16) * KPITCH + 32 * c + 16 * grp];
            v8f ct = v8f{};
            #pragma unroll
            for (int c = 0; c < 8; ++c)
                ct = __builtin_amdgcn_wmma_f32_16x16x32_bf16(false, aq[c], false, bks[c],
                                                             (short)0, ct, false, false);
            lt[h] = ct;
        }

        // ---- streaming softmax + post-softmax gating (C layout: row = r+8*grp, col = ln16) ----
        const float* mrow0 = Mb + (size_t)(q0 + 8 * grp) * GG + kb + ln16;
        float alpha[8];
        float amin = 1.0f;
        __bf16* pw = &ldsP[wv * 16 * 32];
        #pragma unroll
        for (int r = 0; r < 8; ++r) {
            const float l0 = lt[0][r];          // base-2 domain
            const float l1 = lt[1][r];
            float tmax = fmaxf(l0, l1);
            #pragma unroll
            for (int mk = 1; mk <= 8; mk <<= 1) tmax = fmaxf(tmax, __shfl_xor(tmax, mk, 32));
            const float nmax = fmaxf(rowmax[r], tmax);
            const float al   = exp2f(rowmax[r] - nmax);
            rowmax[r] = nmax;
            alpha[r]  = al;
            amin      = fminf(amin, al);
            const float p0 = exp2f(l0 - nmax);
            const float p1 = exp2f(l1 - nmax);
            const float* mr = mrow0 + (size_t)r * GG;
            const float m0 = mr[0];
            const float m1 = mr[16];
            float zs = p0 + p1;
            float ss = p0 * fabsf(m0) + p1 * fabsf(m1);
            #pragma unroll
            for (int mk = 1; mk <= 8; mk <<= 1) {
                zs += __shfl_xor(zs, mk, 32);
                ss += __shfl_xor(ss, mk, 32);
            }
            Zs[r] = Zs[r] * al + zs;
            Ss[r] = Ss[r] * al + ss;
            const int prow = r + 8 * grp;
            pw[prow * 32 + ln16]      = (__bf16)(p0 * m0);
            pw[prow * 32 + 16 + ln16] = (__bf16)(p1 * m1);
        }

        // rescale accumulators only when some row max advanced (alpha==1 exactly otherwise)
        if (__ballot(amin < 1.0f) != 0ULL) {
            #pragma unroll
            for (int t = 0; t < 16; ++t)
                #pragma unroll
                for (int r = 0; r < 8; ++r) acc[t][r] *= alpha[r];
        }

        asm volatile("s_wait_dscnt 0x0" ::: "memory");   // wave-local LDS store->load ordering

        // ---- gated P tile as A-matrix (16x32 bf16) ----
        v16bf a2;
        {
            const v8bf lo = *(const v8bf*)&pw[ln16 * 32 + 16 * grp];
            const v8bf hi = *(const v8bf*)&pw[ln16 * 32 + 8 + 16 * grp];
            #pragma unroll
            for (int e = 0; e < 8; ++e) { a2[e] = lo[e]; a2[8 + e] = hi[e]; }
        }

        // ---- P @ V over 16 N-tiles: batch 4 B-tiles, then 4 back-to-back WMMAs ----
        #pragma unroll
        for (int tg = 0; tg < 4; ++tg) {
            v16bf bvs[4];
            #pragma unroll
            for (int u = 0; u < 4; ++u)
                bvs[u] = *(const v16bf*)&tV[((tg * 4 + u) * 32 + lane) << 4];
            #pragma unroll
            for (int u = 0; u < 4; ++u)
                acc[tg * 4 + u] = __builtin_amdgcn_wmma_f32_16x16x32_bf16(
                    false, a2, false, bvs[u], (short)0, acc[tg * 4 + u], false, false);
        }
    }

    // ---- epilogue: out = O / (S + EPS * Z) ----
    #pragma unroll
    for (int r = 0; r < 8; ++r) {
        const float rinv = 1.0f / (Ss[r] + 1e-8f * Zs[r]);
        float* orow = Ob + (size_t)(q0 + r + 8 * grp) * DD + ln16;
        #pragma unroll
        for (int t = 0; t < 16; ++t) orow[16 * t] = acc[t][r] * rinv;
    }
}

extern "C" void kernel_launch(void* const* d_in, const int* in_sizes, int n_in,
                              void* d_out, int out_size, void* d_ws, size_t ws_size,
                              hipStream_t stream) {
    const float* Q = (const float*)d_in[0];
    const float* K = (const float*)d_in[1];
    const float* V = (const float*)d_in[2];
    const float* M = (const float*)d_in[3];
    float* O = (float*)d_out;
    const int B = in_sizes[0] / (GG * DD);

    __bf16* Kbf = (__bf16*)d_ws;
    __bf16* Vbf = (__bf16*)((char*)d_ws + (size_t)B * GG * KPITCH * 2);

    const size_t elems = (size_t)B * GG * DD;
    prep_kv<<<dim3((unsigned)(elems / 4 / 256)), 256, 0, stream>>>(K, V, Kbf, Vbf);

    dim3 grid(GG / ROWS_PER_WG, B);
    gene_attn_fused<<<grid, 256, 0, stream>>>(Q, Kbf, Vbf, M, O);
}
